// PolicyMultipleMPGNN_35897336660645
// MI455X (gfx1250) — matrix operand order repaired and checked by reading, not compiled, not run on hardware
//
#include <hip/hip_runtime.h>

#define NEG_SLOPE 0.01f

typedef __attribute__((ext_vector_type(16))) _Float16 v16h;
typedef __attribute__((ext_vector_type(8)))  _Float16 h8;
typedef __attribute__((ext_vector_type(4)))  _Float16 h4;
typedef __attribute__((ext_vector_type(8)))  float    v8f;

__device__ __forceinline__ float lrelu(float v) { return v > 0.f ? v : NEG_SLOPE * v; }
__device__ __forceinline__ float4 ld4(const float* p) { return *reinterpret_cast<const float4*>(p); }
__device__ __forceinline__ void st4h(_Float16* p, float4 v) {
  h4 t; t.x = (_Float16)v.x; t.y = (_Float16)v.y; t.z = (_Float16)v.z; t.w = (_Float16)v.w;
  *reinterpret_cast<h4*>(p) = t;
}

// One 16x16 f32 accumulator tile of A(MxK, row-major f16 in LDS, stride lda) x W^T.
// Wt is packed column-major per output: Wt[n*K + k] (f16 in global).
// A fragment: lane (m=l&15, half=l>>4) holds
// K = 16*half + 0..15 (contiguous) within each 32-wide k-step.
// B fragment: lane (n=l&15, half=l>>4) holds K = 16*half + 0..15 (contiguous).
__device__ __forceinline__ v8f wmma_tile(const _Float16* A, int lda,
                                         const _Float16* Wt, int K,
                                         int mt, int nt) {
  const int lane = threadIdx.x & 31;
  const int half = lane >> 4;
  const _Float16* arow = A + (mt * 16 + (lane & 15)) * lda;
  const _Float16* bcol = Wt + (size_t)(nt * 16 + (lane & 15)) * K + 16 * half;
  v8f acc = {};
  for (int k0 = 0; k0 < K; k0 += 32) {
    union { v16h v; h8 h[2]; } a;
    a.h[0] = *reinterpret_cast<const h8*>(arow + k0 + 16 * half);
    a.h[1] = *reinterpret_cast<const h8*>(arow + k0 + 16 * half + 8);
    v16h b = *reinterpret_cast<const v16h*>(bcol + k0);
    acc = __builtin_amdgcn_wmma_f32_16x16x32_f16(false, a.v, false, b, (short)0, acc,
                                                 false, false);
  }
  return acc;
}

// ---------------- utility kernels ----------------

__global__ __launch_bounds__(256) void pack_w_kernel(const float* __restrict__ W,
                                                     _Float16* __restrict__ Wt,
                                                     int I, int O) {
  int idx = blockIdx.x * 256 + threadIdx.x;
  if (idx < I * O) {
    int o = idx / I, i = idx - o * I;
    Wt[o * I + i] = (_Float16)W[i * O + o];
  }
}

__global__ __launch_bounds__(256) void zero_kernel(float* __restrict__ p, int n) {
  int i = blockIdx.x * 256 + threadIdx.x;
  if (i < n) p[i] = 0.f;
}

// dst[n][d] = lrelu(sum_f src[n][f] * W[f][d] + b[d]),  F = 8, D = 64
__global__ __launch_bounds__(256) void embed_kernel(const float* __restrict__ src,
                                                    const float* __restrict__ W,
                                                    const float* __restrict__ b,
                                                    float* __restrict__ dst, int N) {
  int idx = blockIdx.x * 256 + threadIdx.x;
  if (idx < N * 64) {
    int n = idx >> 6, d = idx & 63;
    float acc = b[d];
#pragma unroll
    for (int f = 0; f < 8; ++f) acc += src[n * 8 + f] * W[f * 64 + d];
    dst[idx] = lrelu(acc);
  }
}

// ---------------- fused edge + message kernel ----------------
// Per 128-edge tile:
//   A = [x[row] | x[col] | e | g[batch[row]]]  (128 x 256 f16)
//   e_new = MLP_edge(A);  e += e_new  (in place)
//   m = MLP_node1([x[row] | e_new]);  atomic agg[col] += m;  cnt[col] += 1
__global__ __launch_bounds__(256) void edge_pass_kernel(
    const float* __restrict__ x, float* __restrict__ e, const float* __restrict__ g,
    const int* __restrict__ row, const int* __restrict__ col,
    const int* __restrict__ batch,
    const _Float16* __restrict__ W1t, const float* __restrict__ b1,
    const _Float16* __restrict__ W2t, const float* __restrict__ b2,
    const _Float16* __restrict__ M1t, const float* __restrict__ mb1,
    const _Float16* __restrict__ M2t, const float* __restrict__ mb2,
    float* __restrict__ agg, float* __restrict__ cnt, int E) {
  __shared__ _Float16 A[128][256];
  __shared__ _Float16 H[128][128];
  __shared__ int rowS[128], colS[128], gbS[128];

  const int tid = threadIdx.x;
  const int e0 = blockIdx.x * 128;

  if (tid < 128) {
    int ge = e0 + tid;
    int r = 0, c = 0, gb = 0;
    if (ge < E) {
      r = row[ge]; c = col[ge]; gb = batch[r];
      atomicAdd(&cnt[c], 1.0f);
    }
    rowS[tid] = r; colS[tid] = c; gbS[tid] = gb;
  }
  __syncthreads();

  // gather (float4 granularity: 16 quads per 64-wide section)
  const float4 z4 = make_float4(0.f, 0.f, 0.f, 0.f);
  for (int idx = tid; idx < 128 * 16; idx += 256) {
    int le = idx >> 4, q = idx & 15;
    int ge = e0 + le;
    bool v = ge < E;
    int r = rowS[le], c = colS[le], gb = gbS[le];
    float4 fx = v ? ld4(x + (size_t)r * 64 + q * 4) : z4;
    float4 fc = v ? ld4(x + (size_t)c * 64 + q * 4) : z4;
    float4 fe = v ? ld4(e + (size_t)ge * 64 + q * 4) : z4;
    float4 fg = v ? ld4(g + (size_t)gb * 64 + q * 4) : z4;
    st4h(&A[le][q * 4], fx);
    st4h(&A[le][64 + q * 4], fc);
    st4h(&A[le][128 + q * 4], fe);
    st4h(&A[le][192 + q * 4], fg);
  }
  __syncthreads();

  const int wid = tid >> 5;
  const int lane = tid & 31;
  const int half = lane >> 4;
  const int nl = lane & 15;

  // GEMM1: (128x256) x (256x128) -> H (lrelu)
  for (int t = wid; t < 64; t += 8) {
    int mt = t >> 3, nt = t & 7;
    v8f acc = wmma_tile(&A[0][0], 256, W1t, 256, mt, nt);
    int n = nt * 16 + nl;
    float bb = b1[n];
#pragma unroll
    for (int r2 = 0; r2 < 8; ++r2)
      H[mt * 16 + r2 + 8 * half][n] = (_Float16)lrelu(acc[r2] + bb);
  }
  __syncthreads();

  // GEMM2: (128x128) x (128x64) -> e_new; residual e += e_new; stash e_new f16 in A[:,64:128]
  for (int t = wid; t < 32; t += 8) {
    int mt = t >> 2, nt = t & 3;
    v8f acc = wmma_tile(&H[0][0], 128, W2t, 128, mt, nt);
    int n = nt * 16 + nl;
    float bb = b2[n];
#pragma unroll
    for (int r2 = 0; r2 < 8; ++r2) {
      int m = mt * 16 + r2 + 8 * half;
      float en = acc[r2] + bb;
      A[m][64 + n] = (_Float16)en;
      int ge = e0 + m;
      if (ge < E) e[(size_t)ge * 64 + n] += en;
    }
  }
  __syncthreads();

  // GEMM3: node1 layer1 on [x[row] | e_new] = A[:, 0:128] (lda = 256)
  for (int t = wid; t < 64; t += 8) {
    int mt = t >> 3, nt = t & 7;
    v8f acc = wmma_tile(&A[0][0], 256, M1t, 128, mt, nt);
    int n = nt * 16 + nl;
    float bb = mb1[n];
#pragma unroll
    for (int r2 = 0; r2 < 8; ++r2)
      H[mt * 16 + r2 + 8 * half][n] = (_Float16)lrelu(acc[r2] + bb);
  }
  __syncthreads();

  // GEMM4: messages; scatter-add to agg[col]
  for (int t = wid; t < 32; t += 8) {
    int mt = t >> 2, nt = t & 3;
    v8f acc = wmma_tile(&H[0][0], 128, M2t, 128, mt, nt);
    int n = nt * 16 + nl;
    float bb = mb2[n];
#pragma unroll
    for (int r2 = 0; r2 < 8; ++r2) {
      int m = mt * 16 + r2 + 8 * half;
      int ge = e0 + m;
      if (ge < E) atomicAdd(&agg[(size_t)colS[m] * 64 + n], acc[r2] + bb);
    }
  }
}

// ---------------- node update kernel ----------------
// x_new = MLP_node2([x | agg/cnt | g[batch]]); x += x_new; gsum[batch] += x_new
__global__ __launch_bounds__(256) void node_pass_kernel(
    float* __restrict__ x, const float* __restrict__ g,
    const float* __restrict__ agg, const float* __restrict__ cnt,
    const int* __restrict__ batch,
    const _Float16* __restrict__ W1t, const float* __restrict__ b1,
    const _Float16* __restrict__ W2t, const float* __restrict__ b2,
    float* __restrict__ gsum, float* __restrict__ gcnt, int N) {
  __shared__ _Float16 A[128][192];
  __shared__ _Float16 H[128][128];
  __shared__ int bS[128];

  const int tid = threadIdx.x;
  const int n0 = blockIdx.x * 128;

  if (tid < 128) {
    int gn = n0 + tid;
    int gb = 0;
    if (gn < N) { gb = batch[gn]; atomicAdd(&gcnt[gb], 1.0f); }
    bS[tid] = gb;
  }
  __syncthreads();

  const float4 z4 = make_float4(0.f, 0.f, 0.f, 0.f);
  for (int idx = tid; idx < 128 * 16; idx += 256) {
    int ln = idx >> 4, q = idx & 15;
    int gn = n0 + ln;
    bool v = gn < N;
    int gb = bS[ln];
    float inv = v ? 1.0f / fmaxf(cnt[gn], 1.0f) : 0.f;
    float4 fx = v ? ld4(x + (size_t)gn * 64 + q * 4) : z4;
    float4 fa = v ? ld4(agg + (size_t)gn * 64 + q * 4) : z4;
    float4 fg = v ? ld4(g + (size_t)gb * 64 + q * 4) : z4;
    fa.x *= inv; fa.y *= inv; fa.z *= inv; fa.w *= inv;
    st4h(&A[ln][q * 4], fx);
    st4h(&A[ln][64 + q * 4], fa);
    st4h(&A[ln][128 + q * 4], fg);
  }
  __syncthreads();

  const int wid = tid >> 5;
  const int lane = tid & 31;
  const int half = lane >> 4;
  const int nl = lane & 15;

  for (int t = wid; t < 64; t += 8) {
    int mt = t >> 3, nt = t & 7;
    v8f acc = wmma_tile(&A[0][0], 192, W1t, 192, mt, nt);
    int n = nt * 16 + nl;
    float bb = b1[n];
#pragma unroll
    for (int r2 = 0; r2 < 8; ++r2)
      H[mt * 16 + r2 + 8 * half][n] = (_Float16)lrelu(acc[r2] + bb);
  }
  __syncthreads();

  for (int t = wid; t < 32; t += 8) {
    int mt = t >> 2, nt = t & 3;
    v8f acc = wmma_tile(&H[0][0], 128, W2t, 128, mt, nt);
    int n = nt * 16 + nl;
    float bb = b2[n];
#pragma unroll
    for (int r2 = 0; r2 < 8; ++r2) {
      int m = mt * 16 + r2 + 8 * half;
      int gn = n0 + m;
      if (gn < N) {
        float xn = acc[r2] + bb;
        x[(size_t)gn * 64 + n] += xn;
        atomicAdd(&gsum[(size_t)bS[m] * 64 + n], xn);
      }
    }
  }
}

// ---------------- global update kernel (single block, G <= 64) ----------------
__global__ __launch_bounds__(256) void glob_pass_kernel(
    float* __restrict__ g, const float* __restrict__ gsum,
    const float* __restrict__ gcnt,
    const _Float16* __restrict__ W1t, const float* __restrict__ b1,
    const _Float16* __restrict__ W2t, const float* __restrict__ b2, int G) {
  __shared__ _Float16 A[64][128];
  __shared__ _Float16 H[64][128];
  const int tid = threadIdx.x;
  const float4 z4 = make_float4(0.f, 0.f, 0.f, 0.f);

  for (int idx = tid; idx < 64 * 16; idx += 256) {
    int lg = idx >> 4, q = idx & 15;
    bool v = lg < G;
    float inv = v ? 1.0f / fmaxf(gcnt[lg], 1.0f) : 0.f;
    float4 fg = v ? ld4(g + (size_t)lg * 64 + q * 4) : z4;
    float4 fs = v ? ld4(gsum + (size_t)lg * 64 + q * 4) : z4;
    fs.x *= inv; fs.y *= inv; fs.z *= inv; fs.w *= inv;
    st4h(&A[lg][q * 4], fg);
    st4h(&A[lg][64 + q * 4], fs);
  }
  __syncthreads();

  const int wid = tid >> 5;
  const int lane = tid & 31;
  const int half = lane >> 4;
  const int nl = lane & 15;

  for (int t = wid; t < 32; t += 8) {
    int mt = t >> 3, nt = t & 7;
    v8f acc = wmma_tile(&A[0][0], 128, W1t, 128, mt, nt);
    int n = nt * 16 + nl;
    float bb = b1[n];
#pragma unroll
    for (int r2 = 0; r2 < 8; ++r2)
      H[mt * 16 + r2 + 8 * half][n] = (_Float16)lrelu(acc[r2] + bb);
  }
  __syncthreads();

  for (int t = wid; t < 16; t += 8) {
    int mt = t >> 2, nt = t & 3;
    v8f acc = wmma_tile(&H[0][0], 128, W2t, 128, mt, nt);
    int n = nt * 16 + nl;
    float bb = b2[n];
#pragma unroll
    for (int r2 = 0; r2 < 8; ++r2) {
      int m = mt * 16 + r2 + 8 * half;
      if (m < G) g[(size_t)m * 64 + n] += acc[r2] + bb;
    }
  }
}

// ---------------- decoder ----------------
__global__ __launch_bounds__(256) void decoder_kernel(const float* __restrict__ e,
                                                      const float* __restrict__ wd,
                                                      const float* __restrict__ bd,
                                                      float* __restrict__ out, int E) {
  int i = blockIdx.x * 256 + threadIdx.x;
  if (i < E) {
    float acc = bd[0];
#pragma unroll
    for (int q = 0; q < 16; ++q) {
      float4 v = ld4(e + (size_t)i * 64 + q * 4);
      float4 w = ld4(wd + q * 4);
      acc += v.x * w.x + v.y * w.y + v.z * w.z + v.w * w.w;
    }
    out[i] = acc;
  }
}

// ---------------- launcher ----------------
extern "C" void kernel_launch(void* const* d_in, const int* in_sizes, int n_in,
                              void* d_out, int out_size, void* d_ws, size_t ws_size,
                              hipStream_t stream) {
  (void)n_in; (void)out_size; (void)ws_size;
  const float* x_in  = (const float*)d_in[0];
  const int*   eidx  = (const int*)d_in[1];
  const float* eattr = (const float*)d_in[2];
  const float* u_in  = (const float*)d_in[3];
  const int*   batch = (const int*)d_in[4];
  const int N = in_sizes[0] / 8;
  const int E = in_sizes[2] / 8;
  const int G = in_sizes[3] / 8;
  const int* row = eidx;
  const int* col = eidx + E;

  char* ws = (char*)d_ws;
  size_t off = 0;
  auto alloc = [&](size_t bytes) -> size_t {
    size_t o = off;
    off += (bytes + 255) & ~(size_t)255;
    return o;
  };
  size_t xOff    = alloc((size_t)N * 64 * 4);
  size_t eOff    = alloc((size_t)E * 64 * 4);
  size_t gOff    = alloc((size_t)G * 64 * 4);
  size_t aggOff  = alloc((size_t)N * 64 * 4);
  size_t cntOff  = alloc((size_t)N * 4);
  size_t gsumOff = alloc((size_t)G * 64 * 4);
  size_t gcntOff = alloc((size_t)G * 4);
  size_t zeroEnd = off;
  (void)cntOff; (void)gsumOff; (void)gcntOff;

  float* xbuf = (float*)(ws + xOff);
  float* ebuf = (float*)(ws + eOff);
  float* gbuf = (float*)(ws + gOff);
  float* agg  = (float*)(ws + aggOff);
  float* cnt  = (float*)(ws + cntOff);
  float* gsum = (float*)(ws + gsumOff);
  float* gcnt = (float*)(ws + gcntOff);
  int zn = (int)((zeroEnd - aggOff) / 4);

  // 8 weight matrices per pass: edgeW1, edgeW2, n1W1, n1W2, n2W1, n2W2, gW1, gW2
  static const int dims[8][2] = {{256, 128}, {128, 64}, {128, 128}, {128, 64},
                                 {192, 128}, {128, 64}, {128, 128}, {128, 64}};
  _Float16* pw[3][8];
  for (int p = 0; p < 3; ++p)
    for (int m = 0; m < 8; ++m) {
      size_t o = alloc((size_t)dims[m][0] * dims[m][1] * 2);
      pw[p][m] = (_Float16*)(ws + o);
    }

  // pack weights to f16, [O][I]
  for (int p = 0; p < 3; ++p)
    for (int m = 0; m < 8; ++m) {
      const float* W = (const float*)d_in[13 + p * 16 + m * 2];
      int tot = dims[m][0] * dims[m][1];
      pack_w_kernel<<<(tot + 255) / 256, 256, 0, stream>>>(W, pw[p][m], dims[m][0],
                                                           dims[m][1]);
    }

  // embeddings
  embed_kernel<<<((N * 64) + 255) / 256, 256, 0, stream>>>(
      x_in, (const float*)d_in[5], (const float*)d_in[6], xbuf, N);
  embed_kernel<<<((E * 64) + 255) / 256, 256, 0, stream>>>(
      eattr, (const float*)d_in[7], (const float*)d_in[8], ebuf, E);
  embed_kernel<<<((G * 64) + 255) / 256, 256, 0, stream>>>(
      u_in, (const float*)d_in[9], (const float*)d_in[10], gbuf, G);

  for (int p = 0; p < 3; ++p) {
    const int base = 13 + p * 16;
    zero_kernel<<<(zn + 255) / 256, 256, 0, stream>>>(agg, zn);
    edge_pass_kernel<<<(E + 127) / 128, 256, 0, stream>>>(
        xbuf, ebuf, gbuf, row, col, batch,
        pw[p][0], (const float*)d_in[base + 1], pw[p][1], (const float*)d_in[base + 3],
        pw[p][2], (const float*)d_in[base + 5], pw[p][3], (const float*)d_in[base + 7],
        agg, cnt, E);
    node_pass_kernel<<<(N + 127) / 128, 256, 0, stream>>>(
        xbuf, gbuf, agg, cnt, batch,
        pw[p][4], (const float*)d_in[base + 9], pw[p][5], (const float*)d_in[base + 11],
        gsum, gcnt, N);
    glob_pass_kernel<<<1, 256, 0, stream>>>(
        gbuf, gsum, gcnt,
        pw[p][6], (const float*)d_in[base + 13], pw[p][7],
        (const float*)d_in[base + 15], G);
  }

  decoder_kernel<<<(E + 255) / 256, 256, 0, stream>>>(
      ebuf, (const float*)d_in[11], (const float*)d_in[12], (float*)d_out, E);
}
